// HeteroFloodGNNBenchmark_48086453846276
// MI455X (gfx1250) — matrix operand order repaired and checked by reading, not compiled, not run on hardware
//
#include <hip/hip_runtime.h>

typedef __bf16 bf16_t;
typedef __attribute__((ext_vector_type(16))) __bf16 v16bf;
typedef __attribute__((ext_vector_type(8)))  float  v8f;

constexpr int Hc  = 128;
constexpr int Lc  = 4;
constexpr int N1c = 100000;
constexpr int N2c = 300000;
constexpr int E1c = 300000;
constexpr int E2c = 600000;
constexpr int ECc = 100000;
constexpr int CFc = 6;
constexpr int KPAD = 288;   // 2H + CF = 262 padded to multiple of 32

// ---------------- lane helpers / WMMA fragment loaders (ISA 7.12.2) ----------------
__device__ __forceinline__ int lane_m()  { return threadIdx.x & 15; }
__device__ __forceinline__ int lane_hi() { return (threadIdx.x >> 4) & 1; }

// A fragment 16x32 bf16: lane row = lane&15; pair p -> k = (p/4)*16 + hi*8 + (p%4)*2
__device__ __forceinline__ v16bf load_a_bf16(const bf16_t* __restrict__ row, int kBase) {
  const int hi = lane_hi();
  v16bf a;
#pragma unroll
  for (int p = 0; p < 8; ++p) {
    const int kk = kBase + ((p >> 2) << 4) + (hi << 3) + ((p & 3) << 1);
    a[2 * p]     = row[kk];
    a[2 * p + 1] = row[kk + 1];
  }
  return a;
}

// A fragment from fp32 source with on-the-fly cvt
__device__ __forceinline__ v16bf load_a_f32(const float* __restrict__ row, int kBase) {
  const int hi = lane_hi();
  v16bf a;
#pragma unroll
  for (int p = 0; p < 8; ++p) {
    const int kk = kBase + ((p >> 2) << 4) + (hi << 3) + ((p & 3) << 1);
    a[2 * p]     = (bf16_t)row[kk];
    a[2 * p + 1] = (bf16_t)row[kk + 1];
  }
  return a;
}

// B fragment 32x16 bf16 from weight stored [N x K] row-major (y = x @ W^T):
// lane col n = lane&15; pair p -> k = (p/4)*16 + hi*8 + (p%4)*2 (same as A)
__device__ __forceinline__ v16bf load_b_bf16(const bf16_t* __restrict__ W, int ld,
                                             int col0, int kBase) {
  const int n = lane_m(), hi = lane_hi();
  const bf16_t* row = W + (size_t)(col0 + n) * ld;
  v16bf b;
#pragma unroll
  for (int p = 0; p < 8; ++p) {
    const int kk = kBase + ((p >> 2) << 4) + (hi << 3) + ((p & 3) << 1);
    b[2 * p]     = row[kk];
    b[2 * p + 1] = row[kk + 1];
  }
  return b;
}

__device__ __forceinline__ v8f wmma_bf16(v16bf a, v16bf b, v8f c) {
  return __builtin_amdgcn_wmma_f32_16x16x32_bf16(false, a, false, b, (short)0, c,
                                                 false, false);
}

__device__ __forceinline__ float sigmf(float x) { return 1.f / (1.f + __expf(-x)); }
__device__ __forceinline__ float siluf(float x) { return x / (1.f + __expf(-x)); }

// ---- CDNA5 async global->LDS copy (ASYNCcnt path, ISA ch.10 async / §15.18) ----
__device__ __forceinline__ void async_copy_b128(uint32_t ldsOff, const void* gaddr) {
  asm volatile("global_load_async_to_lds_b128 %0, %1, off"
               :: "v"(ldsOff), "v"((uint64_t)(uintptr_t)gaddr)
               : "memory");
}
__device__ __forceinline__ void wait_async0() {
  asm volatile("s_wait_asynccnt 0x0" ::: "memory");
}
__device__ __forceinline__ uint32_t lds_off(const void* p) {
  return (uint32_t)(uintptr_t)p;   // flat LDS aperture: low 32 bits = LDS offset
}

// ---------------- utility kernels ----------------
__global__ void k_zero(float* __restrict__ p, size_t n) {
  for (size_t i = (size_t)blockIdx.x * blockDim.x + threadIdx.x; i < n;
       i += (size_t)gridDim.x * blockDim.x)
    p[i] = 0.f;
}

__global__ void k_f32_to_bf16(const float* __restrict__ s, bf16_t* __restrict__ d, size_t n) {
  for (size_t i = (size_t)blockIdx.x * blockDim.x + threadIdx.x; i < n;
       i += (size_t)gridDim.x * blockDim.x)
    d[i] = (bf16_t)s[i];
}

// pad/convert weights [rows x cin] fp32 -> [rows x cpad] bf16 (zeros beyond cin)
__global__ void k_pad_w(const float* __restrict__ s, bf16_t* __restrict__ d,
                        int rows, int cin, int cpad) {
  const size_t total = (size_t)rows * cpad;
  for (size_t i = (size_t)blockIdx.x * blockDim.x + threadIdx.x; i < total;
       i += (size_t)gridDim.x * blockDim.x) {
    const int r = (int)(i / cpad), c = (int)(i % cpad);
    d[i] = (c < cin) ? (bf16_t)s[(size_t)r * cin + c] : (bf16_t)0.f;
  }
}

// ---------------- encoder: h = silu(x@W1^T + b1) @ W2^T + b2 ----------------
template <int KIN>
__global__ void k_encoder(const float* __restrict__ x, const float* __restrict__ W1,
                          const float* __restrict__ b1, const bf16_t* __restrict__ W2b,
                          const float* __restrict__ b2, float* __restrict__ h,
                          bf16_t* __restrict__ hb, int N) {
  __shared__ bf16_t hid[16][136];  // padded row stride -> conflict-free A loads
  const int row0 = blockIdx.x * 16;
  // phase 1: tiny K (16/12) first layer in VALU, silu, stash bf16 tile in LDS
  for (int idx = threadIdx.x; idx < 16 * Hc; idx += blockDim.x) {
    const int m = idx >> 7, n = idx & 127;
    const int row = row0 + m;
    float v = 0.f;
    if (row < N) {
      float acc = b1[n];
#pragma unroll
      for (int k = 0; k < KIN; ++k) acc += x[(size_t)row * KIN + k] * W1[n * KIN + k];
      v = siluf(acc);
    }
    hid[m][n] = (bf16_t)v;
  }
  __syncthreads();
  // phase 2: 16x128 WMMA second layer, one 16x16 col-tile per wave
  const int n0 = (threadIdx.x >> 5) * 16;
  v8f c = {0.f, 0.f, 0.f, 0.f, 0.f, 0.f, 0.f, 0.f};
  const bf16_t* aRow = &hid[lane_m()][0];
#pragma unroll
  for (int kb = 0; kb < Hc; kb += 32) {
    v16bf a = load_a_bf16(aRow, kb);
    v16bf b = load_b_bf16(W2b, Hc, n0, kb);
    c = wmma_bf16(a, b, c);
  }
  const int n = lane_m(), hi = lane_hi();
#pragma unroll
  for (int i = 0; i < 8; ++i) {
    const int row = row0 + hi * 8 + i;
    if (row < N) {
      const float v = c[i] + b2[n0 + n];
      h[(size_t)row * Hc + n0 + n]  = v;
      hb[(size_t)row * Hc + n0 + n] = (bf16_t)v;
    }
  }
}

// ------- fused edge kernel: m = [hA[iA[e]], hB[iB[e]], (ec[e])] @ W^T + bias,
//         atomicAdd into acc[iS[e]].
//         A-tile (incl. converted ec tail) staged ONCE per block in LDS:
//         h halves via async global->LDS b128 copies, ec via cvt stores.
//         KP templated (256 / 288) so the K-loop is a straight WMMA chain. -------
template <int KP, bool HAS_EC>
__global__ void k_msg(const int* __restrict__ iA, const int* __restrict__ iB,
                      const int* __restrict__ iS,
                      const bf16_t* __restrict__ hA, const bf16_t* __restrict__ hB,
                      const float* __restrict__ ec,
                      const bf16_t* __restrict__ W, const float* __restrict__ bias,
                      float* __restrict__ acc, int E) {
  constexpr int LDA = KP + 8;        // padded stride: conflict-free LDS A reads
  __shared__ bf16_t aTile[16][LDA];
  const int e0 = blockIdx.x * 16;
  // stage hA|hB halves: 512 x 16B async chunks, 2 per thread
#pragma unroll
  for (int q = 0; q < 2; ++q) {
    const int j   = threadIdx.x + (q << 8);
    const int r   = j >> 5;          // edge row 0..15
    const int sub = j & 31;          // 32 x 16B chunks per row
    const int seg = sub >> 4;        // 0: hA half, 1: hB half
    const int c16 = sub & 15;        // chunk within 256B segment
    const int er  = min(e0 + r, E - 1);
    const bf16_t* src = (seg ? hB + (size_t)iB[er] * Hc
                             : hA + (size_t)iA[er] * Hc) + c16 * 8;
    async_copy_b128(lds_off(&aTile[r][seg * Hc + c16 * 8]), src);
  }
  if (HAS_EC) {  // edge-feature tail: 6 cvt'd floats + zero pad to 32 cols
    for (int idx = threadIdx.x; idx < 16 * 32; idx += 256) {
      const int r = idx >> 5, cc = idx & 31;
      const int er = min(e0 + r, E - 1);
      aTile[r][2 * Hc + cc] =
          (cc < CFc) ? (bf16_t)ec[(size_t)er * CFc + cc] : (bf16_t)0.f;
    }
  }
  wait_async0();
  __syncthreads();

  const int n0 = (threadIdx.x >> 5) * 16;
  const int m  = lane_m();
  v8f c = {0.f, 0.f, 0.f, 0.f, 0.f, 0.f, 0.f, 0.f};
#pragma unroll
  for (int kb = 0; kb < KP; kb += 32) {
    v16bf a = load_a_bf16(&aTile[m][0], kb);
    v16bf b = load_b_bf16(W, KP, n0, kb);
    c = wmma_bf16(a, b, c);
  }
  const int n = lane_m(), hi = lane_hi();
#pragma unroll
  for (int i = 0; i < 8; ++i) {
    const int er = e0 + hi * 8 + i;
    if (er < E) {
      const float v = c[i] + bias[n0 + n];     // per-edge bias, summed by scatter
      unsafeAtomicAdd(&acc[(size_t)iS[er] * Hc + n0 + n], v);
    }
  }
}

// ---------------- fused GRU cell: h = GRU(agg, h) ----------------
// agg (fp32) and h (bf16) tiles async-staged in LDS once (8-24x reuse)
__global__ void k_gru(const float* __restrict__ agg, const bf16_t* __restrict__ hbf,
                      const bf16_t* __restrict__ Wih, const bf16_t* __restrict__ Whh,
                      const float* __restrict__ bih, const float* __restrict__ bhh,
                      float* __restrict__ h, bf16_t* __restrict__ hbOut, int N) {
  __shared__ float  gbuf[2][16][385];  // [gi|gh][row][3H]
  __shared__ float  aggT[16][132];     // staged agg tile (fp32)
  __shared__ bf16_t hbT[16][136];      // staged h tile (bf16)
  const int row0 = blockIdx.x * 16;
  // stage agg: 16 rows x 512B = 512 chunks (2/thread); stage hb: 256 chunks (1/thread)
#pragma unroll
  for (int q = 0; q < 2; ++q) {
    const int j   = threadIdx.x + (q << 8);
    const int r   = j >> 5, cch = j & 31;
    const int row = min(row0 + r, N - 1);
    async_copy_b128(lds_off(&aggT[r][cch * 4]), agg + (size_t)row * Hc + cch * 4);
  }
  {
    const int r = threadIdx.x >> 4, cch = threadIdx.x & 15;
    const int row = min(row0 + r, N - 1);
    async_copy_b128(lds_off(&hbT[r][cch * 8]), hbf + (size_t)row * Hc + cch * 8);
  }
  wait_async0();
  __syncthreads();

  const int wave = threadIdx.x >> 5;
  const int m    = lane_m();
  // 48 WMMA tile-jobs: 24 col-tiles of gi (A=aggT) + 24 of gh (A=hbT)
  for (int job = wave; job < 48; job += 8) {
    const int which = job / 24;        // 0 = gi, 1 = gh (wave-uniform)
    const int n0    = (job % 24) * 16;
    const bf16_t* Wm = which ? Whh : Wih;
    v8f c = {0.f, 0.f, 0.f, 0.f, 0.f, 0.f, 0.f, 0.f};
#pragma unroll
    for (int kb = 0; kb < Hc; kb += 32) {
      v16bf a = which ? load_a_bf16(&hbT[m][0], kb)
                      : load_a_f32(&aggT[m][0], kb);
      v16bf b = load_b_bf16(Wm, Hc, n0, kb);
      c = wmma_bf16(a, b, c);
    }
    const float* bs = which ? bhh : bih;
    const int n = lane_m(), hi = lane_hi();
#pragma unroll
    for (int i = 0; i < 8; ++i) gbuf[which][hi * 8 + i][n0 + n] = c[i] + bs[n0 + n];
  }
  __syncthreads();
  // gate math (torch GRUCell semantics)
  for (int idx = threadIdx.x; idx < 16 * Hc; idx += blockDim.x) {
    const int mm = idx >> 7, n = idx & 127;
    const int row = row0 + mm;
    if (row < N) {
      const float r  = sigmf(gbuf[0][mm][n]       + gbuf[1][mm][n]);
      const float z  = sigmf(gbuf[0][mm][Hc + n]  + gbuf[1][mm][Hc + n]);
      const float nn = tanhf(gbuf[0][mm][2 * Hc + n] + r * gbuf[1][mm][2 * Hc + n]);
      const float ho = h[(size_t)row * Hc + n];
      const float hn = (1.f - z) * nn + z * ho;
      h[(size_t)row * Hc + n]     = hn;
      hbOut[(size_t)row * Hc + n] = (bf16_t)hn;
    }
  }
}

// ---------------- decoder: out = silu(h@W1^T+b1) @ W2^T + b2 (W2 tiny) ----------------
__global__ void k_decoder(const bf16_t* __restrict__ hbf, const bf16_t* __restrict__ W1b,
                          const float* __restrict__ b1, const float* __restrict__ W2,
                          const float* __restrict__ b2, float* __restrict__ out,
                          int N, int OUT) {
  __shared__ float hid[16][68];
  const int row0 = blockIdx.x * 16;
  const int wave = threadIdx.x >> 5;
  if (wave < 4) {  // wave-uniform branch, EXEC stays all-ones inside
    const int n0   = wave * 16;
    const int rowA = min(row0 + lane_m(), N - 1);
    v8f c = {0.f, 0.f, 0.f, 0.f, 0.f, 0.f, 0.f, 0.f};
#pragma unroll
    for (int kb = 0; kb < Hc; kb += 32) {
      v16bf a = load_a_bf16(hbf + (size_t)rowA * Hc, kb);
      v16bf b = load_b_bf16(W1b, Hc, n0, kb);
      c = wmma_bf16(a, b, c);
    }
    const int n = lane_m(), hi = lane_hi();
#pragma unroll
    for (int i = 0; i < 8; ++i)
      hid[hi * 8 + i][n0 + n] = siluf(c[i] + b1[n0 + n]);
  }
  __syncthreads();
  if ((int)threadIdx.x < 16 * OUT) {
    const int mm = threadIdx.x / OUT, o = threadIdx.x % OUT;
    const int row = row0 + mm;
    if (row < N) {
      float a = b2[o];
#pragma unroll
      for (int k = 0; k < Hc / 2; ++k) a += hid[mm][k] * W2[o * (Hc / 2) + k];
      out[(size_t)row * OUT + o] = a;
    }
  }
}

// ---------------- host orchestration ----------------
static inline int ceil16(int n) { return (n + 15) / 16; }

extern "C" void kernel_launch(void* const* d_in, const int* in_sizes, int n_in,
                              void* d_out, int out_size, void* d_ws, size_t ws_size,
                              hipStream_t stream) {
  const float* x1d = (const float*)d_in[0];
  const float* x2d = (const float*)d_in[1];
  const int* ei1 = (const int*)d_in[2];
  const int* ei2 = (const int*)d_in[3];
  const int* eic = (const int*)d_in[4];
  const float* ecf = (const float*)d_in[5];
  const int *s1 = ei1, *t1 = ei1 + E1c;
  const int *s2 = ei2, *t2 = ei2 + E2c;
  const int *sc = eic, *tc = eic + ECc;

  const float* e1W1 = (const float*)d_in[6];  const float* e1b1 = (const float*)d_in[7];
  const float* e1W2 = (const float*)d_in[8];  const float* e1b2 = (const float*)d_in[9];
  const float* e2W1 = (const float*)d_in[10]; const float* e2b1 = (const float*)d_in[11];
  const float* e2W2 = (const float*)d_in[12]; const float* e2b2 = (const float*)d_in[13];
  const float* m1W = (const float*)d_in[14];  const float* m1b = (const float*)d_in[15];
  const float* m2W = (const float*)d_in[16];  const float* m2b = (const float*)d_in[17];
  const float* c12W = (const float*)d_in[18]; const float* c12b = (const float*)d_in[19];
  const float* c21W = (const float*)d_in[20]; const float* c21b = (const float*)d_in[21];
  const float* g1Wih = (const float*)d_in[22]; const float* g1Whh = (const float*)d_in[23];
  const float* g1bih = (const float*)d_in[24]; const float* g1bhh = (const float*)d_in[25];
  const float* g2Wih = (const float*)d_in[26]; const float* g2Whh = (const float*)d_in[27];
  const float* g2bih = (const float*)d_in[28]; const float* g2bhh = (const float*)d_in[29];
  const float* d1W1 = (const float*)d_in[30]; const float* d1b1 = (const float*)d_in[31];
  const float* d1W2 = (const float*)d_in[32]; const float* d1b2 = (const float*)d_in[33];
  const float* d2W1 = (const float*)d_in[34]; const float* d2b1 = (const float*)d_in[35];
  const float* d2W2 = (const float*)d_in[36]; const float* d2b2 = (const float*)d_in[37];
  const float* deW1 = (const float*)d_in[38]; const float* deb1 = (const float*)d_in[39];
  const float* deW2 = (const float*)d_in[40]; const float* deb2 = (const float*)d_in[41];

  // ---- workspace bump allocator ----
  char* wsp = (char*)d_ws;
  auto alloc = [&](size_t bytes) -> void* {
    void* p = (void*)wsp;
    wsp += (bytes + 255) & ~(size_t)255;
    return p;
  };
  float*  h1f = (float*)alloc((size_t)N1c * Hc * 4);
  float*  h2f = (float*)alloc((size_t)N2c * Hc * 4);
  bf16_t* h1b = (bf16_t*)alloc((size_t)N1c * Hc * 2);
  bf16_t* h2b = (bf16_t*)alloc((size_t)N2c * Hc * 2);
  float*  agg = (float*)alloc((size_t)N2c * Hc * 4);  // shared by both graphs
  bf16_t* e1W2b = (bf16_t*)alloc((size_t)Hc * Hc * 2);
  bf16_t* e2W2b = (bf16_t*)alloc((size_t)Hc * Hc * 2);
  bf16_t* m1Wb  = (bf16_t*)alloc((size_t)Lc * Hc * 2 * Hc * 2);
  bf16_t* m2Wb  = (bf16_t*)alloc((size_t)Lc * Hc * 2 * Hc * 2);
  bf16_t* c12Wb = (bf16_t*)alloc((size_t)Lc * Hc * KPAD * 2);
  bf16_t* c21Wb = (bf16_t*)alloc((size_t)Lc * Hc * KPAD * 2);
  bf16_t* g1WihB = (bf16_t*)alloc((size_t)Lc * 3 * Hc * Hc * 2);
  bf16_t* g1WhhB = (bf16_t*)alloc((size_t)Lc * 3 * Hc * Hc * 2);
  bf16_t* g2WihB = (bf16_t*)alloc((size_t)Lc * 3 * Hc * Hc * 2);
  bf16_t* g2WhhB = (bf16_t*)alloc((size_t)Lc * 3 * Hc * Hc * 2);
  bf16_t* d1W1b = (bf16_t*)alloc((size_t)(Hc / 2) * Hc * 2);
  bf16_t* d2W1b = (bf16_t*)alloc((size_t)(Hc / 2) * Hc * 2);
  bf16_t* deW1b = (bf16_t*)alloc((size_t)(Hc / 2) * Hc * 2);
  (void)ws_size; (void)in_sizes; (void)n_in; (void)out_size;

  const dim3 cvtGrid(2048), blk(256);
  // ---- weight conversions (fp32 -> bf16, coupling W zero-padded 262->288) ----
  k_f32_to_bf16<<<cvtGrid, blk, 0, stream>>>(e1W2, e1W2b, (size_t)Hc * Hc);
  k_f32_to_bf16<<<cvtGrid, blk, 0, stream>>>(e2W2, e2W2b, (size_t)Hc * Hc);
  k_f32_to_bf16<<<cvtGrid, blk, 0, stream>>>(m1W, m1Wb, (size_t)Lc * Hc * 2 * Hc);
  k_f32_to_bf16<<<cvtGrid, blk, 0, stream>>>(m2W, m2Wb, (size_t)Lc * Hc * 2 * Hc);
  k_pad_w<<<cvtGrid, blk, 0, stream>>>(c12W, c12Wb, Lc * Hc, 2 * Hc + CFc, KPAD);
  k_pad_w<<<cvtGrid, blk, 0, stream>>>(c21W, c21Wb, Lc * Hc, 2 * Hc + CFc, KPAD);
  k_f32_to_bf16<<<cvtGrid, blk, 0, stream>>>(g1Wih, g1WihB, (size_t)Lc * 3 * Hc * Hc);
  k_f32_to_bf16<<<cvtGrid, blk, 0, stream>>>(g1Whh, g1WhhB, (size_t)Lc * 3 * Hc * Hc);
  k_f32_to_bf16<<<cvtGrid, blk, 0, stream>>>(g2Wih, g2WihB, (size_t)Lc * 3 * Hc * Hc);
  k_f32_to_bf16<<<cvtGrid, blk, 0, stream>>>(g2Whh, g2WhhB, (size_t)Lc * 3 * Hc * Hc);
  k_f32_to_bf16<<<cvtGrid, blk, 0, stream>>>(d1W1, d1W1b, (size_t)(Hc / 2) * Hc);
  k_f32_to_bf16<<<cvtGrid, blk, 0, stream>>>(d2W1, d2W1b, (size_t)(Hc / 2) * Hc);
  k_f32_to_bf16<<<cvtGrid, blk, 0, stream>>>(deW1, deW1b, (size_t)(Hc / 2) * Hc);

  // ---- encoders ----
  k_encoder<16><<<ceil16(N1c), blk, 0, stream>>>(x1d, e1W1, e1b1, e1W2b, e1b2, h1f, h1b, N1c);
  k_encoder<12><<<ceil16(N2c), blk, 0, stream>>>(x2d, e2W1, e2b1, e2W2b, e2b2, h2f, h2b, N2c);

  // ---- message-passing layers ----
  for (int l = 0; l < Lc; ++l) {
    const bf16_t* m1Wl = m1Wb + (size_t)l * Hc * 2 * Hc;
    const bf16_t* m2Wl = m2Wb + (size_t)l * Hc * 2 * Hc;
    const bf16_t* c12Wl = c12Wb + (size_t)l * Hc * KPAD;
    const bf16_t* c21Wl = c21Wb + (size_t)l * Hc * KPAD;
    const bf16_t* g1ihl = g1WihB + (size_t)l * 3 * Hc * Hc;
    const bf16_t* g1hhl = g1WhhB + (size_t)l * 3 * Hc * Hc;
    const bf16_t* g2ihl = g2WihB + (size_t)l * 3 * Hc * Hc;
    const bf16_t* g2hhl = g2WhhB + (size_t)l * 3 * Hc * Hc;

    // graph-1 messages + GRU
    k_zero<<<4096, blk, 0, stream>>>(agg, (size_t)N1c * Hc);
    k_msg<2 * Hc, false><<<ceil16(E1c), blk, 0, stream>>>(
        s1, t1, t1, h1b, h1b, nullptr, m1Wl, m1b + (size_t)l * Hc, agg, E1c);
    k_gru<<<ceil16(N1c), blk, 0, stream>>>(agg, h1b, g1ihl, g1hhl,
                                           g1bih + (size_t)l * 3 * Hc,
                                           g1bhh + (size_t)l * 3 * Hc, h1f, h1b, N1c);
    // graph-2 messages + GRU
    k_zero<<<4096, blk, 0, stream>>>(agg, (size_t)N2c * Hc);
    k_msg<2 * Hc, false><<<ceil16(E2c), blk, 0, stream>>>(
        s2, t2, t2, h2b, h2b, nullptr, m2Wl, m2b + (size_t)l * Hc, agg, E2c);
    k_gru<<<ceil16(N2c), blk, 0, stream>>>(agg, h2b, g2ihl, g2hhl,
                                           g2bih + (size_t)l * 3 * Hc,
                                           g2bhh + (size_t)l * 3 * Hc, h2f, h2b, N2c);
    // coupling 1d->2d: m = [h1[sc], h2[tc], ec] @ c12W^T + b; h2 += scatter@tc
    k_msg<KPAD, true><<<ceil16(ECc), blk, 0, stream>>>(
        sc, tc, tc, h1b, h2b, ecf, c12Wl, c12b + (size_t)l * Hc, h2f, ECc);
    k_f32_to_bf16<<<cvtGrid, blk, 0, stream>>>(h2f, h2b, (size_t)N2c * Hc);
    // coupling 2d->1d (uses updated h2): m = [h2[tc], h1[sc], ec] @ c21W^T + b; h1 += scatter@sc
    k_msg<KPAD, true><<<ceil16(ECc), blk, 0, stream>>>(
        tc, sc, sc, h2b, h1b, ecf, c21Wl, c21b + (size_t)l * Hc, h1f, ECc);
    k_f32_to_bf16<<<cvtGrid, blk, 0, stream>>>(h1f, h1b, (size_t)N1c * Hc);
  }

  // ---- decoders: d_out = [o1 (N1x2) | o2 (N2x1) | oe (N1x2)] ----
  float* o1 = (float*)d_out;
  float* o2 = o1 + (size_t)N1c * 2;
  float* oe = o2 + (size_t)N2c;
  k_decoder<<<ceil16(N1c), blk, 0, stream>>>(h1b, d1W1b, d1b1, d1W2, d1b2, o1, N1c, 2);
  k_decoder<<<ceil16(N2c), blk, 0, stream>>>(h2b, d2W1b, d2b1, d2W2, d2b2, o2, N2c, 1);
  k_decoder<<<ceil16(N1c), blk, 0, stream>>>(h1b, deW1b, deb1, deW2, deb2, oe, N1c, 2);
}
